// Llama4Model_87222195847677
// MI455X (gfx1250) — matrix operand order repaired and checked by reading, not compiled, not run on hardware
//
#include <hip/hip_runtime.h>
#include <hip/hip_bf16.h>

#define SEQ   512
#define BATCH 2
#define NTOK  (SEQ * BATCH)   // 1024 tokens
#define HDIM  1024
#define IDIM  2816
#define NEXP  8

typedef __attribute__((ext_vector_type(16))) __bf16 v16bf;
typedef __attribute__((ext_vector_type(8)))  float  v8f;

union BFrag { unsigned u[8]; uint4 q[2]; v16bf v; };

// fp32 pair -> packed bf16 (a in [15:0], b in [31:16]), RNE, single VALU op.
// Builtin is not declared for gfx1250 on this toolchain; emit the CDNA
// instruction directly.
__device__ __forceinline__ unsigned pack2_bf16(float a, float b) {
  unsigned r;
  asm("v_cvt_pk_bf16_f32 %0, %1, %2" : "=v"(r) : "v"(a), "v"(b));
  return r;
}

// pack 16 consecutive K floats (64B) into 8 bf16-pair dwords of a B fragment
__device__ __forceinline__ void pack_b_frag(BFrag& b, const float* __restrict__ p) {
  const float4* q = (const float4*)p;
  const float4 f0 = q[0], f1 = q[1], f2 = q[2], f3 = q[3];
  b.u[0] = pack2_bf16(f0.x, f0.y); b.u[1] = pack2_bf16(f0.z, f0.w);
  b.u[2] = pack2_bf16(f1.x, f1.y); b.u[3] = pack2_bf16(f1.z, f1.w);
  b.u[4] = pack2_bf16(f2.x, f2.y); b.u[5] = pack2_bf16(f2.z, f2.w);
  b.u[6] = pack2_bf16(f3.x, f3.y); b.u[7] = pack2_bf16(f3.z, f3.w);
}

// ---------------- Kernel 1: router logits + argmax + bf16 pack of hidden ----
__global__ __launch_bounds__(256) void k_router_pack(
    const float* __restrict__ hs, const float* __restrict__ rw,
    int* __restrict__ routing, unsigned* __restrict__ xbf)
{
  const int t   = blockIdx.x;
  const int tid = threadIdx.x;
  __shared__ float red[256 * NEXP];

  const float4 x = ((const float4*)(hs + (size_t)t * HDIM))[tid];
  #pragma unroll
  for (int e = 0; e < NEXP; ++e) {
    const float4 w = ((const float4*)(rw + (size_t)e * HDIM))[tid];
    red[tid * NEXP + e] = x.x * w.x + x.y * w.y + x.z * w.z + x.w * w.w;
  }
  __syncthreads();
  for (int s = 128; s > 0; s >>= 1) {
    if (tid < s) {
      #pragma unroll
      for (int e = 0; e < NEXP; ++e)
        red[tid * NEXP + e] += red[(tid + s) * NEXP + e];
    }
    __syncthreads();
  }
  if (tid == 0) {
    int best = 0; float bv = red[0];
    #pragma unroll
    for (int e = 1; e < NEXP; ++e) {
      float v = red[e];
      if (v > bv) { bv = v; best = e; }   // strict '>' == first-max (jnp.argmax)
    }
    routing[t] = best;
  }
  uint2 px;
  px.x = pack2_bf16(x.x, x.y);
  px.y = pack2_bf16(x.z, x.w);
  *(uint2*)(xbf + (size_t)t * (HDIM / 2) + tid * 2) = px;
}

// ---------------- Kernel 2: deterministic per-expert compaction -------------
__global__ void k_compact(const int* __restrict__ routing,
                          int* __restrict__ counts, int* __restrict__ list)
{
  const int e = threadIdx.x;
  if (e >= NEXP) return;
  int c = 0;
  for (int t = 0; t < NTOK; ++t)
    if (routing[t] == e) list[e * NTOK + (c++)] = t;
  counts[e] = c;
}

// ---------------- Kernel 3: gate & up GEMMs + SiLU (bf16 WMMA, f32 acc) -----
__global__ __launch_bounds__(256) void k_gate_up(
    const float* __restrict__ gw, const float* __restrict__ uw,
    const unsigned* __restrict__ xbf,
    const int* __restrict__ counts, const int* __restrict__ list,
    unsigned short* __restrict__ act)
{
  const int e   = blockIdx.z;
  const int cnt = counts[e];
  const int m0  = blockIdx.y * 16;
  if (m0 >= cnt) return;                       // uniform early exit

  const int lane = threadIdx.x & 31;
  const int wave = threadIdx.x >> 5;
  const int half = lane >> 4;
  const int l16  = lane & 15;
  const int n    = blockIdx.x * 128 + wave * 16 + l16;   // column of I

  int mr = m0 + l16; if (mr >= cnt) mr = cnt - 1;        // clamp (masked later)
  const int tokA = list[e * NTOK + mr];
  const unsigned* arow = xbf + (size_t)tokA * (HDIM / 2) + 8 * half;
  const float*    grow = gw + ((size_t)e * IDIM + n) * HDIM + 16 * half;
  const float*    urow = uw + ((size_t)e * IDIM + n) * HDIM + 16 * half;

  v8f accG = {}; v8f accU = {};
  #pragma unroll 2
  for (int kb = 0; kb < HDIM; kb += 32) {
    BFrag a, bg, bu;
    // A 16x32 bf16: per lane 32B contiguous K-run at kb + 16*half
    a.q[0] = *(const uint4*)(arow + (kb >> 1));
    a.q[1] = *(const uint4*)(arow + (kb >> 1) + 4);
    // B 32x16 bf16: per lane 64B contiguous K-run at kb + 16*half
    pack_b_frag(bg, grow + kb);
    pack_b_frag(bu, urow + kb);
    __builtin_prefetch(grow + kb + 256, 0, 0);   // 1KB ahead -> global_prefetch
    __builtin_prefetch(urow + kb + 256, 0, 0);
    accG = __builtin_amdgcn_wmma_f32_16x16x32_bf16(false, a.v, false, bg.v,
                                                   (short)0, accG, false, false);
    accU = __builtin_amdgcn_wmma_f32_16x16x32_bf16(false, a.v, false, bu.v,
                                                   (short)0, accU, false, false);
  }
  #pragma unroll
  for (int v = 0; v < 8; ++v) {
    const int m = m0 + v + 8 * half;           // C/D: M = v + 8*(lane>>4)
    if (m < cnt) {
      const float g = accG[v], u = accU[v];
      const float sig = __builtin_amdgcn_rcpf(1.0f + __expf(-g));
      const float aval = g * sig * u;          // silu(g) * u
      const int tok = list[e * NTOK + m];
      act[(size_t)tok * IDIM + n] = (unsigned short)pack2_bf16(aval, 0.0f);
    }
  }
}

// ---------------- Kernel 4: down GEMM + scatter to output --------------------
__global__ __launch_bounds__(256) void k_down(
    const float* __restrict__ dw,
    const unsigned* __restrict__ actu,
    const int* __restrict__ counts, const int* __restrict__ list,
    float* __restrict__ out)
{
  const int e   = blockIdx.z;
  const int cnt = counts[e];
  const int m0  = blockIdx.y * 16;
  if (m0 >= cnt) return;

  const int lane = threadIdx.x & 31;
  const int wave = threadIdx.x >> 5;
  const int half = lane >> 4;
  const int l16  = lane & 15;
  const int n    = blockIdx.x * 128 + wave * 16 + l16;   // h index

  int mr = m0 + l16; if (mr >= cnt) mr = cnt - 1;
  const int tokA = list[e * NTOK + mr];
  const unsigned* arow = actu + (size_t)tokA * (IDIM / 2) + 8 * half;
  const float*    wrow = dw + ((size_t)e * HDIM + n) * IDIM + 16 * half;

  v8f acc = {};
  #pragma unroll 2
  for (int kb = 0; kb < IDIM; kb += 32) {
    BFrag a, b;
    a.q[0] = *(const uint4*)(arow + (kb >> 1));
    a.q[1] = *(const uint4*)(arow + (kb >> 1) + 4);
    pack_b_frag(b, wrow + kb);
    __builtin_prefetch(wrow + kb + 256, 0, 0);
    acc = __builtin_amdgcn_wmma_f32_16x16x32_bf16(false, a.v, false, b.v,
                                                  (short)0, acc, false, false);
  }
  #pragma unroll
  for (int v = 0; v < 8; ++v) {
    const int m = m0 + v + 8 * half;
    if (m < cnt) {
      const int tok = list[e * NTOK + m];
      out[(size_t)tok * HDIM + n] = acc[v];    // every token written exactly once
    }
  }
}

// ---------------- launch ----------------------------------------------------
extern "C" void kernel_launch(void* const* d_in, const int* in_sizes, int n_in,
                              void* d_out, int out_size, void* d_ws, size_t ws_size,
                              hipStream_t stream) {
  const float* hs = (const float*)d_in[0];   // [S,B,H]
  const float* rw = (const float*)d_in[1];   // [E,H]
  const float* gw = (const float*)d_in[2];   // [E,I,H]
  const float* uw = (const float*)d_in[3];   // [E,I,H]
  const float* dw = (const float*)d_in[4];   // [E,H,I]
  float* out = (float*)d_out;

  char* ws = (char*)d_ws;
  int*            routing = (int*)(ws + 0);                      // 4 KB
  int*            counts  = (int*)(ws + 4096);                   // 32 B
  int*            list    = (int*)(ws + 8192);                   // 32 KB
  unsigned*       xbf     = (unsigned*)(ws + 65536);             // 2 MB bf16 hidden
  unsigned short* act     = (unsigned short*)(ws + 65536 + 2 * 1024 * 1024); // 5.5 MB

  k_router_pack<<<dim3(NTOK), dim3(256), 0, stream>>>(hs, rw, routing, xbf);
  k_compact<<<dim3(1), dim3(32), 0, stream>>>(routing, counts, list);
  k_gate_up<<<dim3(IDIM / 128, NTOK / 16, NEXP), dim3(256), 0, stream>>>(
      gw, uw, xbf, counts, list, act);
  k_down<<<dim3(HDIM / 128, NTOK / 16, NEXP), dim3(256), 0, stream>>>(
      dw, (const unsigned*)act, counts, list, out);
}
